// TrajectoryDecoder_81947976008223
// MI455X (gfx1250) — matrix operand order repaired and checked by reading, not compiled, not run on hardware
//
#include <hip/hip_runtime.h>
#include <hip/hip_bf16.h>
#include <math.h>

typedef unsigned int u32;
typedef __attribute__((ext_vector_type(8)))  float  v8f;
typedef __attribute__((ext_vector_type(16))) __bf16 v16bf;
typedef __attribute__((ext_vector_type(8)))  __bf16 v8bf;
typedef __attribute__((ext_vector_type(4)))  u32    v4u;
typedef __attribute__((ext_vector_type(8)))  int    v8i;
typedef __attribute__((ext_vector_type(4)))  int    v4i;

// RNE f32 -> bf16 via bit trick
__device__ __forceinline__ __bf16 f2bf(float f) {
    unsigned u = __builtin_bit_cast(unsigned, f);
    unsigned r = (u + 0x7FFFu + ((u >> 16) & 1u)) >> 16;
    unsigned short s = (unsigned short)r;
    return __builtin_bit_cast(__bf16, s);
}

#define LDSK 80   // bf16 elems per LDS row: 160 B stride (32 B multiple)

// ---------------------------------------------------------------------------
// TDM: 2D bf16 tile (tile_k x tile_rows) -> LDS with 160B row padding.
// D# per ISA 8.3/8.4. Row stride in elements. Tiles always in-bounds, so
// tensor dims == tile dims. pad: 8 DWORDs every 32 DWORDs -> LDSK=80 stride.
// ---------------------------------------------------------------------------
#if __has_builtin(__builtin_amdgcn_tensor_load_to_lds)
#define HAVE_TDM 1
__device__ __forceinline__ void tdm_load_tile_bf16(
    u32 lds_off, const __bf16* gptr, u32 tile_k, u32 tile_rows, u32 row_stride)
{
    unsigned long long ga = (unsigned long long)(size_t)gptr;
    v4u g0;
    g0[0] = 1u;                                           // count=1
    g0[1] = lds_off;                                      // lds_addr
    g0[2] = (u32)(ga & 0xFFFFFFFFu);                      // global_addr[31:0]
    g0[3] = (u32)((ga >> 32) & 0x1FFFFFFu) | (2u << 30);  // addr hi | type=2
    u32 w0 = (1u << 16)   // data_size = 1 (2 bytes)
           | (1u << 20)   // pad_enable
           | (4u << 22)   // pad_interval: 32 DWORDs (one 64-elem bf16 row)
           | (7u << 25);  // pad_amount: 8 DWORDs (16 bf16) -> stride 80 elems
    unsigned long long d0s = row_stride;                  // tensor_dim0_stride
    v8i g1;
    g1[0] = (int)w0;                                      // mask=0 | sizes/pads
    g1[1] = (int)((tile_k & 0xFFFFu) << 16);              // tensor_dim0 lo
    g1[2] = (int)(((tile_k >> 16) & 0xFFFFu) | ((tile_rows & 0xFFFFu) << 16));
    g1[3] = (int)(((tile_rows >> 16) & 0xFFFFu) | ((tile_k & 0xFFFFu) << 16)); // tile_dim0
    g1[4] = (int)(tile_rows & 0xFFFFu);                   // tile_dim1, tile_dim2=0
    g1[5] = (int)(d0s & 0xFFFFFFFFu);
    g1[6] = (int)((d0s >> 32) & 0xFFFFu);                 // dim1_stride unused (2D)
    g1[7] = 0;
    v4i g2 = {0, 0, 0, 0};
    v4i g3 = {0, 0, 0, 0};
#if defined(__clang_major__) && __clang_major__ >= 23
    v8i g4 = {0, 0, 0, 0, 0, 0, 0, 0};
    __builtin_amdgcn_tensor_load_to_lds(g0, g1, g2, g3, g4, 0);
#else
    __builtin_amdgcn_tensor_load_to_lds(g0, g1, g2, g3, 0);
#endif
}
#else
#define HAVE_TDM 0
#endif

// ---------------------------------------------------------------------------
// 1) AdaptiveAvgPool2d(8): single streaming pass over 2.15 GB (HBM-bound leg)
// ---------------------------------------------------------------------------
__global__ __launch_bounds__(64) void pool_kernel(const float* __restrict__ bev,
                                                  float* __restrict__ pooled) {
    int bc = blockIdx.x;                       // b*512 + c
    const float* src = bev + (size_t)bc * 4096;
    int t  = threadIdx.x;                      // 0..63
    int py = t >> 3, px = t & 7;
    const float* rowp = src + py * 8 * 64 + px * 8;
    float s = 0.f;
#pragma unroll
    for (int r = 0; r < 8; ++r) {
        float4 a = *(const float4*)(rowp);
        float4 b = *(const float4*)(rowp + 4);
        s += a.x + a.y + a.z + a.w + b.x + b.y + b.z + b.w;
        rowp += 64;
    }
    pooled[(size_t)bc * 64 + t] = s * (1.f / 64.f);
}

// ---------------------------------------------------------------------------
// Shared WMMA compute over 64x64 LDS tiles (stride LDSK), wave tile 32x32.
// ---------------------------------------------------------------------------
__device__ __forceinline__ void wmma_tile_compute(
    const __bf16* As, const __bf16* Bs, int lane, int m0, int n0, v8f acc[2][2])
{
#pragma unroll
    for (int kk = 0; kk < 64; kk += 32) {
        v16bf afrag[2], bfrag[2];
#pragma unroll
        for (int i = 0; i < 2; ++i) {
            int r  = m0 + i * 16 + (lane & 15);
            int kb = kk + ((lane >> 4) * 16);
            v8bf lo = *(const v8bf*)&As[r * LDSK + kb];
            v8bf hi = *(const v8bf*)&As[r * LDSK + kb + 8];
            afrag[i] = __builtin_shufflevector(lo, hi,
                0, 1, 2, 3, 4, 5, 6, 7, 8, 9, 10, 11, 12, 13, 14, 15);
            int cidx = n0 + i * 16 + (lane & 15);
            int kb2  = kk + ((lane >> 4) * 16);
            bfrag[i] = *(const v16bf*)&Bs[cidx * LDSK + kb2];
        }
#pragma unroll
        for (int i = 0; i < 2; ++i)
#pragma unroll
            for (int j = 0; j < 2; ++j)
                acc[i][j] = __builtin_amdgcn_wmma_f32_16x16x32_bf16(
                    false, afrag[i], false, bfrag[j],
                    (short)0, acc[i][j], false, false);
    }
}

// ---------------------------------------------------------------------------
// 2) Pool GEMM (runs once): f32 operands, on-the-fly bf16 convert, split-K.
//    Writes raw partials C[z*M*N + ...]; reduction kernel applies epilogue.
// ---------------------------------------------------------------------------
__global__ __launch_bounds__(128) void gemm_f32src_wmma(
    const float* __restrict__ A, int lda,
    const float* __restrict__ W, int ldb,
    float* __restrict__ C, int M, int N, int K)
{
    __shared__ __bf16 As[64 * LDSK];
    __shared__ __bf16 Bs[64 * LDSK];
    const int tid  = threadIdx.x;
    const int lane = tid & 31;
    const int wave = tid >> 5;
    const int m0 = (wave >> 1) * 32;
    const int n0 = (wave & 1) * 32;
    const int rowBase = blockIdx.y * 64;
    const int colBase = blockIdx.x * 64;
    const int kPerSplit = K / gridDim.z;
    const int kBeg = blockIdx.z * kPerSplit;
    const int kEnd = kBeg + kPerSplit;

    v8f acc[2][2] = {};
    const int lr = tid >> 3;
    const int lc = (tid & 7) * 8;

    for (int k0 = kBeg; k0 < kEnd; k0 += 64) {
#pragma unroll
        for (int p = 0; p < 4; ++p) {
            int row = p * 16 + lr;
            const float* ga = A + (size_t)(rowBase + row) * lda + k0 + lc;
            float4 a0 = *(const float4*)ga;
            float4 a1 = *(const float4*)(ga + 4);
            __bf16* da = &As[row * LDSK + lc];
            da[0] = f2bf(a0.x); da[1] = f2bf(a0.y); da[2] = f2bf(a0.z); da[3] = f2bf(a0.w);
            da[4] = f2bf(a1.x); da[5] = f2bf(a1.y); da[6] = f2bf(a1.z); da[7] = f2bf(a1.w);
            const float* gb = W + (size_t)(colBase + row) * ldb + k0 + lc;
            float4 b0 = *(const float4*)gb;
            float4 b1 = *(const float4*)(gb + 4);
            __bf16* db = &Bs[row * LDSK + lc];
            db[0] = f2bf(b0.x); db[1] = f2bf(b0.y); db[2] = f2bf(b0.z); db[3] = f2bf(b0.w);
            db[4] = f2bf(b1.x); db[5] = f2bf(b1.y); db[6] = f2bf(b1.z); db[7] = f2bf(b1.w);
        }
        if (k0 + 64 < kEnd) {   // warm GL2 for next K tile (global_prefetch_b8)
            __builtin_prefetch(A + (size_t)(rowBase + lr) * lda + k0 + 64 + lc, 0, 1);
            __builtin_prefetch(W + (size_t)(colBase + lr) * ldb + k0 + 64 + lc, 0, 1);
        }
        __syncthreads();
        wmma_tile_compute(As, Bs, lane, m0, n0, acc);
        __syncthreads();
    }

    const int colOff = lane & 15;
    const int rowOff = (lane >> 4) * 8;
#pragma unroll
    for (int i = 0; i < 2; ++i)
#pragma unroll
        for (int j = 0; j < 2; ++j)
#pragma unroll
            for (int v = 0; v < 8; ++v) {
                int row = rowBase + m0 + i * 16 + rowOff + v;
                int col = colBase + n0 + j * 16 + colOff;
                C[((size_t)blockIdx.z * M + row) * N + col] = acc[i][j][v];
            }
}

// ---------------------------------------------------------------------------
// 3) GRU GEMM (runs 61x): pure-bf16 operands, tiles fetched by the TDM.
//    C[M,N] = A[M,K]bf16 @ W[N,K]bf16^T + bias, f32 out.
// ---------------------------------------------------------------------------
__global__ __launch_bounds__(128) void gru_gemm_tdm(
    const __bf16* __restrict__ A,     // M x K, row-major, lda = K
    const __bf16* __restrict__ W,     // N x K, row-major, ldb = K
    const float* __restrict__ bias,
    float* __restrict__ C, int M, int N, int K)
{
    __shared__ __bf16 As[64 * LDSK];
    __shared__ __bf16 Bs[64 * LDSK];
    const int tid  = threadIdx.x;
    const int lane = tid & 31;
    const int wave = tid >> 5;
    const int m0 = (wave >> 1) * 32;
    const int n0 = (wave & 1) * 32;
    const int rowBase = blockIdx.y * 64;
    const int colBase = blockIdx.x * 64;

    v8f acc[2][2] = {};

    for (int k0 = 0; k0 < K; k0 += 64) {
#if HAVE_TDM
        if (wave == 0) {    // one wave programs the Tensor Data Mover
            tdm_load_tile_bf16((u32)(size_t)(&As[0]),
                               A + (size_t)rowBase * K + k0, 64, 64, (u32)K);
            tdm_load_tile_bf16((u32)(size_t)(&Bs[0]),
                               W + (size_t)colBase * K + k0, 64, 64, (u32)K);
            __builtin_amdgcn_s_wait_tensorcnt(0);
        }
#else
        {   // fallback: cooperative 32-bf16-per-thread copies
            int lr = tid >> 2, lc = (tid & 3) * 16;
#pragma unroll
            for (int p = 0; p < 2; ++p) {
                int row = p * 32 + lr;
                *(v16bf*)&As[row * LDSK + lc] =
                    *(const v16bf*)(A + (size_t)(rowBase + row) * K + k0 + lc);
                *(v16bf*)&Bs[row * LDSK + lc] =
                    *(const v16bf*)(W + (size_t)(colBase + row) * K + k0 + lc);
            }
        }
#endif
        __syncthreads();
        wmma_tile_compute(As, Bs, lane, m0, n0, acc);
        __syncthreads();
    }

    const int colOff = lane & 15;
    const int rowOff = (lane >> 4) * 8;
#pragma unroll
    for (int i = 0; i < 2; ++i)
#pragma unroll
        for (int j = 0; j < 2; ++j)
#pragma unroll
            for (int v = 0; v < 8; ++v) {
                int row = rowBase + m0 + i * 16 + rowOff + v;
                int col = colBase + n0 + j * 16 + colOff;
                C[(size_t)row * N + col] = acc[i][j][v] + bias[col];
            }
}

// ---------------------------------------------------------------------------
// 4) Split-K reduce + pool epilogue: featsb = bf16(relu(sum+pool_b)+language)
// ---------------------------------------------------------------------------
__global__ __launch_bounds__(256) void pool_epilogue_kernel(
    const float* __restrict__ part, const float* __restrict__ bias,
    const float* __restrict__ addend, __bf16* __restrict__ outb,
    int MN, int N, int S)
{
    int idx = blockIdx.x * 256 + threadIdx.x;
    if (idx >= MN) return;
    float s = 0.f;
    for (int z = 0; z < S; ++z) s += part[(size_t)z * MN + idx];
    s += bias[idx % N];
    outb[idx] = f2bf(fmaxf(s, 0.f) + addend[idx]);
}

// ---------------------------------------------------------------------------
// 5) Weight f32 -> bf16 repack (once per call; GRU weights are L2-resident)
// ---------------------------------------------------------------------------
__global__ __launch_bounds__(256) void cvt_bf16_kernel(
    const float* __restrict__ src, int src_ld,
    __bf16* __restrict__ dst, int rows, int cols)
{
    int idx = blockIdx.x * 256 + threadIdx.x;
    if (idx >= rows * cols) return;
    int r = idx / cols, c = idx % cols;
    dst[idx] = f2bf(src[(size_t)r * src_ld + c]);
}

// ---------------------------------------------------------------------------
// 6) GRU cell: h = (1-z)*n + z*h_prev; writes f32 and bf16 copies of h.
//    Layer 0 folds the rank-2 prev-waypoint term of w_ih0[:,512:514].
// ---------------------------------------------------------------------------
__global__ __launch_bounds__(256) void gru_cell_kernel(
    const float* __restrict__ gi, const float* __restrict__ gh,
    float* __restrict__ h, __bf16* __restrict__ hb,
    const float* __restrict__ wih, int wih_ld,     // nullable (layer 1)
    const float* __restrict__ traj, int t,
    int B, int H)
{
    int idx = blockIdx.x * blockDim.x + threadIdx.x;
    if (idx >= B * H) return;
    int b = idx / H, j = idx % H;
    const float* gib = gi + (size_t)b * 3 * H;
    const float* ghb = gh + (size_t)b * 3 * H;
    float ir = gib[j], iz = gib[H + j], in = gib[2 * H + j];
    if (wih) {
        float wx = 0.f, wy = 0.f;
        if (t > 0) {
            wx = traj[((size_t)b * 20 + (t - 1)) * 2 + 0];
            wy = traj[((size_t)b * 20 + (t - 1)) * 2 + 1];
        }
        ir += wx * wih[(size_t)(j)         * wih_ld + 512] + wy * wih[(size_t)(j)         * wih_ld + 513];
        iz += wx * wih[(size_t)(H + j)     * wih_ld + 512] + wy * wih[(size_t)(H + j)     * wih_ld + 513];
        in += wx * wih[(size_t)(2 * H + j) * wih_ld + 512] + wy * wih[(size_t)(2 * H + j) * wih_ld + 513];
    }
    float r = 1.f / (1.f + __expf(-(ir + ghb[j])));
    float z = 1.f / (1.f + __expf(-(iz + ghb[H + j])));
    float n = tanhf(in + r * ghb[2 * H + j]);
    float hn = (1.f - z) * n + z * h[idx];
    h[idx]  = hn;
    hb[idx] = f2bf(hn);
}

// ---------------------------------------------------------------------------
// 7) Heads: one 256-thread block per batch row.
// ---------------------------------------------------------------------------
__global__ __launch_bounds__(256) void heads_kernel(
    const float* __restrict__ h1,
    const float* __restrict__ w1, const float* __restrict__ b1,
    const float* __restrict__ w2, const float* __restrict__ b2,
    const float* __restrict__ spw, const float* __restrict__ spb,
    const float* __restrict__ hdw, const float* __restrict__ hdb,
    float* __restrict__ out, int t)
{
    __shared__ float sh[512];
    __shared__ float shid[256];
    __shared__ float s0[256], s1[256], s2[256], s3[256];
    int b = blockIdx.x, j = threadIdx.x;
    sh[j]       = h1[(size_t)b * 512 + j];
    sh[j + 256] = h1[(size_t)b * 512 + 256 + j];
    __syncthreads();
    float acc = b1[j];
#pragma unroll 4
    for (int k = 0; k < 512; ++k) acc += sh[k] * w1[(size_t)j * 512 + k];
    shid[j] = fmaxf(acc, 0.f);
    __syncthreads();
    s0[j] = shid[j] * w2[j];
    s1[j] = shid[j] * w2[256 + j];
    s2[j] = sh[j] * spw[j] + sh[j + 256] * spw[j + 256];
    s3[j] = sh[j] * hdw[j] + sh[j + 256] * hdw[j + 256];
    __syncthreads();
    for (int s = 128; s > 0; s >>= 1) {
        if (j < s) { s0[j] += s0[j + s]; s1[j] += s1[j + s];
                     s2[j] += s2[j + s]; s3[j] += s3[j + s]; }
        __syncthreads();
    }
    if (j == 0) {
        out[((size_t)b * 20 + t) * 2 + 0] = s0[0] + b2[0];
        out[((size_t)b * 20 + t) * 2 + 1] = s1[0] + b2[1];
        out[10240 + (size_t)b * 20 + t]   = s2[0] + spb[0];
        out[15360 + (size_t)b * 20 + t]   = s3[0] + hdb[0];
    }
}

__global__ __launch_bounds__(256) void zero_kernel(float* p, int n) {
    int i = blockIdx.x * 256 + threadIdx.x;
    if (i < n) p[i] = 0.f;
}

// ---------------------------------------------------------------------------
// Orchestration
// ---------------------------------------------------------------------------
extern "C" void kernel_launch(void* const* d_in, const int* in_sizes, int n_in,
                              void* d_out, int out_size, void* d_ws, size_t ws_size,
                              hipStream_t stream)
{
    (void)in_sizes; (void)n_in; (void)out_size; (void)ws_size;
    const float* bev  = (const float*)d_in[0];
    const float* lang = (const float*)d_in[1];
    const float* pw   = (const float*)d_in[2];
    const float* pb   = (const float*)d_in[3];
    const float* wih0 = (const float*)d_in[4];
    const float* whh0 = (const float*)d_in[5];
    const float* bih0 = (const float*)d_in[6];
    const float* bhh0 = (const float*)d_in[7];
    const float* wih1 = (const float*)d_in[8];
    const float* whh1 = (const float*)d_in[9];
    const float* bih1 = (const float*)d_in[10];
    const float* bhh1 = (const float*)d_in[11];
    const float* wpw1 = (const float*)d_in[12];
    const float* wpb1 = (const float*)d_in[13];
    const float* wpw2 = (const float*)d_in[14];
    const float* wpb2 = (const float*)d_in[15];
    const float* spw  = (const float*)d_in[16];
    const float* spb  = (const float*)d_in[17];
    const float* hdw  = (const float*)d_in[18];
    const float* hdb  = (const float*)d_in[19];
    float* out = (float*)d_out;

    char* wsb = (char*)d_ws;
    float*  pooled = (float*)wsb;  wsb += (size_t)256 * 32768 * 4;  // 33.5 MB
    float*  cpart  = (float*)wsb;  wsb += (size_t)8 * 256 * 512 * 4;
    float*  gi0    = (float*)wsb;  wsb += (size_t)256 * 1536 * 4;
    float*  gh0    = (float*)wsb;  wsb += (size_t)256 * 1536 * 4;
    float*  gi1    = (float*)wsb;  wsb += (size_t)256 * 1536 * 4;
    float*  gh1    = (float*)wsb;  wsb += (size_t)256 * 1536 * 4;
    float*  h0     = (float*)wsb;  wsb += (size_t)256 * 512 * 4;    // h0,h1
    float*  h1     = (float*)wsb;  wsb += (size_t)256 * 512 * 4;    //  contiguous
    __bf16* hb0    = (__bf16*)wsb; wsb += (size_t)256 * 512 * 2;    // hb0,hb1
    __bf16* hb1    = (__bf16*)wsb; wsb += (size_t)256 * 512 * 2;    //  contiguous
    __bf16* featsb = (__bf16*)wsb; wsb += (size_t)256 * 512 * 2;
    __bf16* whh0b  = (__bf16*)wsb; wsb += (size_t)1536 * 512 * 2;
    __bf16* wih1b  = (__bf16*)wsb; wsb += (size_t)1536 * 512 * 2;
    __bf16* whh1b  = (__bf16*)wsb; wsb += (size_t)1536 * 512 * 2;
    __bf16* wih0cb = (__bf16*)wsb; wsb += (size_t)1536 * 512 * 2;

    // init hidden states (f32 + bf16) every call; ws is not re-zeroed by harness
    zero_kernel<<<(2 * 256 * 512) / 256, 256, 0, stream>>>(h0, 2 * 256 * 512);
    zero_kernel<<<(256 * 512) / 256, 256, 0, stream>>>((float*)hb0, 256 * 512);

    // one-time bf16 repack of L2-resident GRU weights
    const int CVT = 1536 * 512;
    cvt_bf16_kernel<<<CVT / 256, 256, 0, stream>>>(whh0, 512, whh0b, 1536, 512);
    cvt_bf16_kernel<<<CVT / 256, 256, 0, stream>>>(wih1, 512, wih1b, 1536, 512);
    cvt_bf16_kernel<<<CVT / 256, 256, 0, stream>>>(whh1, 512, whh1b, 1536, 512);
    cvt_bf16_kernel<<<CVT / 256, 256, 0, stream>>>(wih0, 514, wih0cb, 1536, 512);

    // streaming avg-pool (HBM-bound)
    pool_kernel<<<256 * 512, 64, 0, stream>>>(bev, pooled);

    // pool linear: (256 x 32768) @ (32768 x 512), split-K x8 for occupancy
    gemm_f32src_wmma<<<dim3(512 / 64, 256 / 64, 8), 128, 0, stream>>>(
        pooled, 32768, pw, 32768, cpart, 256, 512, 32768);
    pool_epilogue_kernel<<<(256 * 512 + 255) / 256, 256, 0, stream>>>(
        cpart, pb, lang, featsb, 256 * 512, 512, 8);

    // time-invariant layer-0 input gates: feats @ w_ih0[:, :512]^T + b_ih0
    gru_gemm_tdm<<<dim3(1536 / 64, 256 / 64, 1), 128, 0, stream>>>(
        featsb, wih0cb, bih0, gi0, 256, 1536, 512);

    for (int t = 0; t < 20; ++t) {
        gru_gemm_tdm<<<dim3(24, 4, 1), 128, 0, stream>>>(
            hb0, whh0b, bhh0, gh0, 256, 1536, 512);
        gru_cell_kernel<<<512, 256, 0, stream>>>(
            gi0, gh0, h0, hb0, wih0, 514, out, t, 256, 512);
        gru_gemm_tdm<<<dim3(24, 4, 1), 128, 0, stream>>>(
            hb0, wih1b, bih1, gi1, 256, 1536, 512);
        gru_gemm_tdm<<<dim3(24, 4, 1), 128, 0, stream>>>(
            hb1, whh1b, bhh1, gh1, 256, 1536, 512);
        gru_cell_kernel<<<512, 256, 0, stream>>>(
            gi1, gh1, h1, hb1, nullptr, 0, nullptr, t, 256, 512);
        heads_kernel<<<256, 256, 0, stream>>>(
            h1, wpw1, wpb1, wpw2, wpb2, spw, spb, hdw, hdb, out, t);
    }
}
